// AttnBlock_58806692217557
// MI455X (gfx1250) — compile-verified, matrix operand order checked by reading, not run on hardware
//
#include <hip/hip_runtime.h>
#include <hip/hip_bf16.h>

// ---------------------------------------------------------------------------
// Problem constants
// ---------------------------------------------------------------------------
#define NPIX   8192          // B*H*W = 8*32*32
#define CDIM   768
#define QKVD   2304
#define OUTD   256
#define KDIM   768           // GEMM reduction dim (always 768 here)

typedef __attribute__((ext_vector_type(16))) __bf16 v16bf;
typedef __attribute__((ext_vector_type(8)))  __bf16 v8bf;
typedef __attribute__((ext_vector_type(8)))  float  v8f;
typedef __attribute__((ext_vector_type(4)))  int    v4i;

typedef __attribute__((address_space(1))) v4i g_v4i;   // global int4
typedef __attribute__((address_space(3))) v4i l_v4i;   // LDS int4

static __device__ __forceinline__ int iclamp(int v, int lo, int hi) {
    return v < lo ? lo : (v > hi ? hi : v);
}

// ---------------------------------------------------------------------------
// CDNA5 async global->LDS copy (16B per lane), ASYNCcnt-tracked.
// Falls back to a synchronous vector copy if the builtin is unavailable.
// ---------------------------------------------------------------------------
#if __has_builtin(__builtin_amdgcn_global_load_async_to_lds_b128)
#define HAS_ASYNC_LDS 1
#else
#define HAS_ASYNC_LDS 0
#endif

static __device__ __forceinline__ void copy16_g2l(const void* g, void* l) {
#if HAS_ASYNC_LDS
    __builtin_amdgcn_global_load_async_to_lds_b128(
        (g_v4i*)(uintptr_t)g, (l_v4i*)(uintptr_t)l, 0, 0);
#else
    *(uint4*)l = *(const uint4*)g;
#endif
}
static __device__ __forceinline__ void async_drain() {
#if HAS_ASYNC_LDS
    asm volatile("s_wait_asynccnt 0x0" ::: "memory");
#endif
}

// Load one 16x32 bf16 WMMA fragment from a row-major matrix (row ptr given).
// Lane layout per CDNA5 ISA 7.12.2:
//   lanes 0-15 : K = kb+0..7   and K = kb+16..23
//   lanes 16-31: K = kb+8..15  and K = kb+24..31
static __device__ __forceinline__ v16bf load_frag(const __bf16* rowp, int kb, int hi8) {
    const v8bf lo = *(const v8bf*)(rowp + kb + hi8);
    const v8bf hp = *(const v8bf*)(rowp + kb + 16 + hi8);
    v16bf f;
#pragma unroll
    for (int i = 0; i < 8; ++i) { f[i] = lo[i]; f[i + 8] = hp[i]; }
    return f;
}

static __device__ __forceinline__ v16bf combine_frag(v8bf lo, v8bf hp) {
    v16bf f;
#pragma unroll
    for (int i = 0; i < 8; ++i) { f[i] = lo[i]; f[i + 8] = hp[i]; }
    return f;
}

// ---------------------------------------------------------------------------
// Weight convert+transpose: dst[n*K + k] = bf16(src[k*N + n])
// ---------------------------------------------------------------------------
__global__ void k_transpose_bf16(const float* __restrict__ src,
                                 __bf16* __restrict__ dst, int K, int N) {
    int idx = blockIdx.x * 256 + threadIdx.x;
    if (idx >= K * N) return;
    int n = idx % N, k = idx / N;                  // consecutive threads: coalesced src read
    dst[(size_t)n * K + k] = (__bf16)src[(size_t)k * N + n];
}

// ---------------------------------------------------------------------------
// tbar[n,c] = 3x3 replicate-pad box mean of x  (== mean_kk t[n,kk,c])
// ---------------------------------------------------------------------------
__global__ void k_tbar(const float* __restrict__ x, float* __restrict__ tbar) {
    int idx = blockIdx.x * 256 + threadIdx.x;
    if (idx >= NPIX * CDIM) return;
    int n = idx & (NPIX - 1), c = idx >> 13;       // n fastest -> coalesced x reads along W
    int b = n >> 10, hw = n & 1023, h = hw >> 5, w = hw & 31;
    const float* xp = x + ((size_t)b * CDIM + c) * 1024;
    float s = 0.f;
#pragma unroll
    for (int di = -1; di <= 1; ++di) {
        int hh = iclamp(h + di, 0, 31);
#pragma unroll
        for (int dj = -1; dj <= 1; ++dj) {
            int ww = iclamp(w + dj, 0, 31);
            s += xp[hh * 32 + ww];
        }
    }
    tbar[(size_t)n * CDIM + c] = s * (1.f / 9.f);
}

// ---------------------------------------------------------------------------
// LN1: per-pixel layernorm of x (B,C,H,W) -> hn (NPIX, 768) bf16.
// LDS tile of 16 pixels x 768 channels so global reads are coalesced along W.
// ---------------------------------------------------------------------------
__global__ void k_ln1(const float* __restrict__ x, const float* __restrict__ g,
                      const float* __restrict__ be, __bf16* __restrict__ hn) {
    __shared__ __align__(16) float s[16][CDIM];    // 48 KB
    const int tid = threadIdx.x;
    const int pixBase = blockIdx.x * 16;           // 1024 % 16 == 0 -> single image
    const int b = pixBase >> 10, hw = pixBase & 1023;
    const float* xb = x + (size_t)b * CDIM * 1024 + hw;
    for (int idx = tid; idx < CDIM * 16; idx += 256) {
        int c = idx >> 4, p = idx & 15;            // p fastest -> coalesced
        s[p][c] = xb[(size_t)c * 1024 + p];
    }
    __syncthreads();
    const int lane = tid & 31, wave = tid >> 5;
#pragma unroll
    for (int pi = 0; pi < 2; ++pi) {
        int p = wave * 2 + pi;
        const float* row = s[p];
        float sum = 0.f, sq = 0.f;
        for (int i = lane; i < CDIM; i += 32) { float v = row[i]; sum += v; sq += v * v; }
#pragma unroll
        for (int o = 16; o; o >>= 1) {
            sum += __shfl_xor(sum, o, 32);
            sq  += __shfl_xor(sq,  o, 32);
        }
        float mu  = sum * (1.f / CDIM);
        float var = sq * (1.f / CDIM) - mu * mu;
        float rs  = rsqrtf(var + 1e-5f);
        __bf16* dst = hn + (size_t)(pixBase + p) * CDIM;
        for (int i = lane; i < CDIM; i += 32)
            dst[i] = (__bf16)((row[i] - mu) * rs * g[i] + be[i]);
    }
}

// ---------------------------------------------------------------------------
// LN2: wave-per-row layernorm of contiguous fp32 y -> bf16 z (coalesced).
// ---------------------------------------------------------------------------
__global__ void k_ln2(const float* __restrict__ y, const float* __restrict__ g,
                      const float* __restrict__ be, __bf16* __restrict__ z) {
    const int lane = threadIdx.x & 31, wave = threadIdx.x >> 5;
    const int row = blockIdx.x * 8 + wave;
    const float* r = y + (size_t)row * CDIM;
    float sum = 0.f, sq = 0.f;
    for (int i = lane; i < CDIM; i += 32) { float v = r[i]; sum += v; sq += v * v; }
#pragma unroll
    for (int o = 16; o; o >>= 1) {
        sum += __shfl_xor(sum, o, 32);
        sq  += __shfl_xor(sq,  o, 32);
    }
    float mu  = sum * (1.f / CDIM);
    float var = sq * (1.f / CDIM) - mu * mu;
    float rs  = rsqrtf(var + 1e-5f);
    __bf16* dst = z + (size_t)row * CDIM;
    for (int i = lane; i < CDIM; i += 32)
        dst[i] = (__bf16)((r[i] - mu) * rs * g[i] + be[i]);
}

// ---------------------------------------------------------------------------
// WMMA GEMM:  C(MxN) = A(MxK,bf16) * B^T  (B stored as (N,K) bf16 rows) + bias
//   MODE 0: store bf16 row-major (qkv)
//   MODE 1: addend[m*N+n] = acc + bias + addend[m*N+n]   (y = tbar + proj)
//   MODE 2: relu(acc+bias) -> out[(b*OUTD+col)*1024 + hw] (final NCHW store)
// Block = 8 waves, tile 128(M) x 64(N).
// B panel (64 x 768) staged once in LDS via async global->LDS (shared by all
// 8 waves -> 8x less L2 traffic); rows padded to 776 elems to spread banks.
// A is register double-buffered so WMMA issue is decoupled from LOADcnt.
// ---------------------------------------------------------------------------
template <int MODE>
__global__ void k_gemm(const __bf16* __restrict__ A, const __bf16* __restrict__ B,
                       const float* __restrict__ bias, float* __restrict__ addend,
                       void* __restrict__ out, int M, int N) {
    constexpr int BROW = 776;                          // padded LDS row stride
    __shared__ __align__(16) __bf16 sB[64 * BROW];     // ~97 KB
    const int tid = threadIdx.x;
    const int nBase = blockIdx.y * 64;

    // ---- stage B panel: 64 rows x 768 cols = 6144 16B chunks, 24/thread ----
    {
        const __bf16* bsrc = B + (size_t)nBase * KDIM;
        for (int idx = tid; idx < 64 * 96; idx += 256) {
            int row = idx / 96, off = (idx % 96) * 8;  // off in bf16 elems
            copy16_g2l(bsrc + (size_t)row * KDIM + off, sB + row * BROW + off);
        }
        async_drain();
    }
    __syncthreads();

    const int lane = tid & 31, wave = tid >> 5;
    const int l2 = lane & 15, mh = lane >> 4, hi8 = mh * 8;
    const int mBase = blockIdx.x * 128 + wave * 16;
    const __bf16* arow = A + (size_t)(mBase + l2) * KDIM;
    const __bf16* brow[4];
#pragma unroll
    for (int j = 0; j < 4; ++j)
        brow[j] = sB + (16 * j + l2) * BROW;

    v8f acc[4] = {};
    v8bf alo = *(const v8bf*)(arow + hi8);
    v8bf ahi = *(const v8bf*)(arow + 16 + hi8);
#pragma unroll 4
    for (int kb = 0; kb < KDIM; kb += 32) {
        v8bf nlo = {}, nhi = {};
        if (kb + 32 < KDIM) {                          // prefetch next A fragment
            nlo = *(const v8bf*)(arow + kb + 32 + hi8);
            nhi = *(const v8bf*)(arow + kb + 48 + hi8);
        }
        v16bf a = combine_frag(alo, ahi);
#pragma unroll
        for (int j = 0; j < 4; ++j) {
            v16bf b = load_frag(brow[j], kb, hi8);     // ds_load_b128 x2
            acc[j] = __builtin_amdgcn_wmma_f32_16x16x32_bf16(
                false, a, false, b, (short)0, acc[j], false, false);
        }
        alo = nlo; ahi = nhi;
    }

#pragma unroll
    for (int j = 0; j < 4; ++j) {
        const int col = nBase + 16 * j + l2;
        const float bv = bias[col];
#pragma unroll
        for (int r = 0; r < 8; ++r) {
            const int m = mBase + r + 8 * mh;          // C/D layout: VGPR r, lane half
            float v = acc[j][r] + bv;
            if (MODE == 0) {
                ((__bf16*)out)[(size_t)m * N + col] = (__bf16)v;
            } else if (MODE == 1) {
                size_t idx = (size_t)m * N + col;
                addend[idx] = v + addend[idx];
            } else {
                v = v > 0.f ? v : 0.f;
                int bb = m >> 10, hw = m & 1023;
                ((float*)out)[((size_t)bb * OUTD + col) * 1024 + hw] = v;
            }
        }
    }
}

// ---------------------------------------------------------------------------
// Attention: one block per pixel, one wave per head.
// Async-stages 9 neighbor qkv rows (bf16) in LDS, computes S=QK^T/sqrt(d),
// softmax over k, column-means over q, then obar = colmean(attn) @ V.
// ---------------------------------------------------------------------------
__global__ void k_attn(const __bf16* __restrict__ qkv, __bf16* __restrict__ obar) {
    __shared__ __align__(16) __bf16 sq[9][QKVD];   // 41.5 KB
    __shared__ float sS[8][96];                    // 81 scores + 9 colmeans
    const int n = blockIdx.x;
    const int b = n >> 10, hw = n & 1023, h = hw >> 5, w = hw & 31;
    const int tid = threadIdx.x;

    int p[9];
#pragma unroll
    for (int kk = 0; kk < 9; ++kk) {
        int hh = iclamp(h + kk / 3 - 1, 0, 31);
        int ww = iclamp(w + kk % 3 - 1, 0, 31);
        p[kk] = (b << 10) + hh * 32 + ww;
    }
    // async 16B-chunk copy of 9 rows x 4608 B
    for (int idx = tid; idx < 9 * 288; idx += 256) {
        int kk = idx / 288, off = (idx % 288) * 8;
        copy16_g2l(qkv + (size_t)p[kk] * QKVD + off, &sq[kk][off]);
    }
    async_drain();
    __syncthreads();

    const int head = tid >> 5, lane = tid & 31;
    const __bf16* Q  = &sq[0][head * 96];
    const __bf16* Kp = &sq[0][768 + head * 96];
    const __bf16* V  = &sq[0][1536 + head * 96];
    const float scale = 0.1020620726159658f;       // 96^-0.5

    for (int e = lane; e < 81; e += 32) {
        const __bf16* qr = Q  + (e / 9) * QKVD;
        const __bf16* kr = Kp + (e % 9) * QKVD;
        float s = 0.f;
#pragma unroll 8
        for (int d = 0; d < 96; ++d) s += (float)qr[d] * (float)kr[d];
        sS[head][e] = s * scale;
    }
    __syncthreads();

    if (lane < 9) {                                // softmax over k, row q = lane
        float* row = &sS[head][lane * 9];
        float m = row[0];
#pragma unroll
        for (int k = 1; k < 9; ++k) m = fmaxf(m, row[k]);
        float e[9], sum = 0.f;
#pragma unroll
        for (int k = 0; k < 9; ++k) { e[k] = __expf(row[k] - m); sum += e[k]; }
        float inv = 1.f / sum;
#pragma unroll
        for (int k = 0; k < 9; ++k) row[k] = e[k] * inv;
    }
    __syncthreads();

    if (lane < 9) {                                // column means over q, k = lane
        float s = 0.f;
#pragma unroll
        for (int q = 0; q < 9; ++q) s += sS[head][q * 9 + lane];
        sS[head][81 + lane] = s * (1.f / 9.f);     // disjoint from [0..80]
    }
    __syncthreads();

    for (int d = lane; d < 96; d += 32) {          // obar = colmean @ V
        float s = 0.f;
#pragma unroll
        for (int k = 0; k < 9; ++k) s += sS[head][81 + k] * (float)V[k * QKVD + d];
        obar[(size_t)n * CDIM + head * 96 + d] = (__bf16)s;
    }
}

// ---------------------------------------------------------------------------
// Host launcher
// ---------------------------------------------------------------------------
extern "C" void kernel_launch(void* const* d_in, const int* in_sizes, int n_in,
                              void* d_out, int out_size, void* d_ws, size_t ws_size,
                              hipStream_t stream) {
    const float* x      = (const float*)d_in[0];
    const float* w_qkv  = (const float*)d_in[1];
    const float* b_qkv  = (const float*)d_in[2];
    const float* w_proj = (const float*)d_in[3];
    const float* b_proj = (const float*)d_in[4];
    const float* g1     = (const float*)d_in[5];
    const float* beta1  = (const float*)d_in[6];
    const float* g2     = (const float*)d_in[7];
    const float* beta2  = (const float*)d_in[8];
    const float* w_fc   = (const float*)d_in[9];
    const float* b_fc   = (const float*)d_in[10];
    float* out = (float*)d_out;

    char* ws = (char*)d_ws;
    // workspace carve-up (all offsets 256B aligned), ~93 MB total
    __bf16* hn      = (__bf16*)(ws);                       // 12.58 MB  (also z)
    __bf16* wT_qkv  = (__bf16*)(ws + 12582912);            //  3.54 MB
    __bf16* wT_proj = (__bf16*)(ws + 16121856);            //  1.18 MB
    __bf16* wT_fc   = (__bf16*)(ws + 17301504);            //  0.39 MB
    __bf16* qkv     = (__bf16*)(ws + 17694720);            // 37.75 MB
    __bf16* obar    = (__bf16*)(ws + 55443456);            // 12.58 MB
    float*  tbar    = (float*) (ws + 68026368);            // 25.17 MB  (also y)

    // 1) weights -> bf16, transposed to (N,K)
    k_transpose_bf16<<<(KDIM * QKVD + 255) / 256, 256, 0, stream>>>(w_qkv, wT_qkv, KDIM, QKVD);
    k_transpose_bf16<<<(KDIM * CDIM + 255) / 256, 256, 0, stream>>>(w_proj, wT_proj, KDIM, CDIM);
    k_transpose_bf16<<<(KDIM * OUTD + 255) / 256, 256, 0, stream>>>(w_fc, wT_fc, KDIM, OUTD);

    // 2) 3x3 box mean (mean_kk of t)
    k_tbar<<<(NPIX * CDIM) / 256, 256, 0, stream>>>(x, tbar);

    // 3) per-pixel LN1 -> hn bf16
    k_ln1<<<NPIX / 16, 256, 0, stream>>>(x, g1, beta1, hn);

    // 4) qkv = hn @ w_qkv + b_qkv   (8192x768 x 768x2304, WMMA bf16)
    k_gemm<0><<<dim3(NPIX / 128, QKVD / 64), 256, 0, stream>>>(
        hn, wT_qkv, b_qkv, nullptr, (void*)qkv, NPIX, QKVD);

    // 5) attention column-mean path -> obar bf16
    k_attn<<<NPIX, 256, 0, stream>>>(qkv, obar);

    // 6) y = tbar + obar @ w_proj + b_proj  (in-place into tbar buffer)
    k_gemm<1><<<dim3(NPIX / 128, CDIM / 64), 256, 0, stream>>>(
        obar, wT_proj, b_proj, tbar, nullptr, NPIX, CDIM);

    // 7) LN2 -> z bf16 (reuse hn buffer)
    k_ln2<<<NPIX / 8, 256, 0, stream>>>(tbar, g2, beta2, hn);

    // 8) out = relu(z @ w_fc + b_fc), stored NCHW
    k_gemm<2><<<dim3(NPIX / 128, OUTD / 64), 256, 0, stream>>>(
        hn, wT_fc, b_fc, nullptr, (void*)out, NPIX, OUTD);
}
